// Encoder_26104811225237
// MI455X (gfx1250) — matrix operand order repaired and checked by reading, not compiled, not run on hardware
//
#include <hip/hip_runtime.h>
#include <hip/hip_bf16.h>

// ---------------------------------------------------------------------------
// Point-group encoder for MI455X (gfx1250, wave32, WMMA).
//   K0: weight convert fp32 -> f16, transposed [N][K] (B-matrix friendly)
//   K1: FPS (1 workgroup per batch, dists in registers, u64 argmax in LDS)
//   K2: KNN top-64 (dists in 256KB LDS, 64 rounds of u64 argmin)
//   K3: gather + 4x WMMA MLP layers (templated K/N, A-frags register-
//       resident, weight prefetch) + maxpool + 512->60 projection
// ---------------------------------------------------------------------------

typedef _Float16 f16_t;
typedef __attribute__((ext_vector_type(16))) _Float16 v16h;
typedef __attribute__((ext_vector_type(8)))  _Float16 v8h;
typedef __attribute__((ext_vector_type(8)))  float    v8f;

#define NPTS 65536
#define NGRP 512
#define GSZ  64
#define EMB  60

// ---------------------------------------------------------------------------
// K0: convert W [K][N] fp32 -> WT [N][Kpad] f16 (zero-padded along K)
// ---------------------------------------------------------------------------
__global__ void cvt_wT(const float* __restrict__ W, f16_t* __restrict__ WT,
                       int K, int N, int Kpad) {
  int i = blockIdx.x * blockDim.x + threadIdx.x;
  if (i >= N * Kpad) return;
  int n = i / Kpad, k = i % Kpad;
  WT[i] = (k < K) ? (f16_t)W[k * N + n] : (f16_t)0.f;
}

// ---------------------------------------------------------------------------
// K1: farthest point sampling.  grid=B, block=1024.  Each thread owns 64
// points; running min-distance kept in registers (fully unrolled).
// Argmax via packed u64 key (dist bits high, ~idx low for first-index ties).
// ---------------------------------------------------------------------------
__global__ void fps_kernel(const float* __restrict__ xyz,
                           float* __restrict__ centers) {
  const int b   = blockIdx.x;
  const int tid = threadIdx.x;
  __shared__ float cpos[3];
  __shared__ unsigned long long red[1024];

  const float* base = xyz + (size_t)b * NPTS * 3;
  float dist[64];
#pragma unroll
  for (int j = 0; j < 64; ++j) dist[j] = 1e10f;

  if (tid == 0) {
    cpos[0] = base[0]; cpos[1] = base[1]; cpos[2] = base[2];
    float* c0 = centers + (size_t)b * NGRP * 3;
    c0[0] = base[0]; c0[1] = base[1]; c0[2] = base[2];
  }
  __syncthreads();

  for (int it = 1; it < NGRP; ++it) {
    const float cx = cpos[0], cy = cpos[1], cz = cpos[2];
    unsigned long long best = 0ull;
#pragma unroll
    for (int j = 0; j < 64; ++j) {
      const int p = tid + (j << 10);
      const float* q = base + (size_t)p * 3;
      const float dx = q[0] - cx, dy = q[1] - cy, dz = q[2] - cz;
      float d = dx * dx + dy * dy + dz * dz;
      d = d < dist[j] ? d : dist[j];
      dist[j] = d;
      unsigned long long key =
          ((unsigned long long)__float_as_uint(d) << 32) |
          (unsigned long long)(0xFFFFFFFFu - (unsigned)p);
      best = key > best ? key : best;
    }
    red[tid] = best;
    __syncthreads();
    for (int s = 512; s > 0; s >>= 1) {
      if (tid < s) {
        unsigned long long o = red[tid + s];
        if (o > red[tid]) red[tid] = o;
      }
      __syncthreads();
    }
    if (tid == 0) {
      const unsigned w = 0xFFFFFFFFu - (unsigned)red[0];
      const float* q = base + (size_t)w * 3;
      cpos[0] = q[0]; cpos[1] = q[1]; cpos[2] = q[2];
      float* co = centers + ((size_t)b * NGRP + it) * 3;
      co[0] = q[0]; co[1] = q[1]; co[2] = q[2];
    }
    __syncthreads();
  }
}

// ---------------------------------------------------------------------------
// K2: KNN top-64.  grid = B*NGRP, block=256, dynamic LDS = 256KB dists +
// 2KB reduce.  64 rounds of argmin with key = (dist_bits<<32)|idx
// (min prefers the lower index on ties, matching top_k stability).
// ---------------------------------------------------------------------------
__global__ void knn_kernel(const float* __restrict__ xyz,
                           const float* __restrict__ centers,
                           int* __restrict__ knn) {
  extern __shared__ char smem[];
  float* dls = (float*)smem;                                         // [65536]
  unsigned long long* red = (unsigned long long*)(smem + NPTS * 4);  // [256]

  const int g   = blockIdx.x;
  const int b   = g >> 9;
  const int tid = threadIdx.x;
  const float cx = centers[(size_t)g * 3 + 0];
  const float cy = centers[(size_t)g * 3 + 1];
  const float cz = centers[(size_t)g * 3 + 2];
  const float* base = xyz + (size_t)b * NPTS * 3;

  for (int p = tid; p < NPTS; p += 256) {
    const float* q = base + (size_t)p * 3;
    const float dx = q[0] - cx, dy = q[1] - cy, dz = q[2] - cz;
    dls[p] = dx * dx + dy * dy + dz * dz;
  }
  __syncthreads();

  for (int r = 0; r < GSZ; ++r) {
    unsigned long long best = ~0ull;
    for (int p = tid; p < NPTS; p += 256) {
      unsigned long long key =
          ((unsigned long long)__float_as_uint(dls[p]) << 32) |
          (unsigned long long)(unsigned)p;
      if (key < best) best = key;
    }
    red[tid] = best;
    __syncthreads();
    for (int s = 128; s > 0; s >>= 1) {
      if (tid < s) {
        unsigned long long o = red[tid + s];
        if (o < red[tid]) red[tid] = o;
      }
      __syncthreads();
    }
    if (tid == 0) {
      const unsigned w = (unsigned)red[0];
      knn[(size_t)g * GSZ + r] = (int)w;
      dls[w] = __uint_as_float(0x7F800000u);  // +inf, key stays monotonic
    }
    __syncthreads();
  }
}

// ---------------------------------------------------------------------------
// WMMA MLP layer: Xout[64][N] = relu(Xin[64][K] @ W[K][N] + bias).
// Xin/Xout in LDS row-major; WT in global as [N][K] (transposed weights).
// 8 waves; wave w owns row-tile mt = w&3 (constant!), so its A fragments for
// ALL K-steps are loaded once into registers and reused for every col tile.
// Fragment layouts per CDNA5 ISA 7.12.2 (f16 A 16x32, f32 C/D 16x16).
// ---------------------------------------------------------------------------
template <int K, int N>
__device__ __forceinline__ void mlp_layer(const f16_t* __restrict__ Xin,
                                          const f16_t* __restrict__ WT,
                                          const float* __restrict__ bias,
                                          f16_t* __restrict__ Xout) {
  constexpr int KSTEPS = K / 32;
  constexpr int NT = 4 * (N / 16);   // 4 row-tiles x N/16 col-tiles
  const int tid  = threadIdx.x;
  const int wave = tid >> 5;
  const int lane = tid & 31;
  const int l16  = lane & 15;
  const int hi   = lane >> 4;        // 0: lanes 0-15, 1: lanes 16-31
  const int mt   = wave & 3;         // fixed row tile for this wave

  // A: lane holds row l16 of tile mt; K runs {kh..kh+7, kh+16..kh+23},
  // kh = hi*8.  Register-resident across all output col tiles.
  v16h afrag[KSTEPS];
  {
    const f16_t* arow = Xin + ((mt << 4) + l16) * K + hi * 8;
#pragma unroll
    for (int s = 0; s < KSTEPS; ++s) {
      v8h a0 = *(const v8h*)(arow + s * 32);
      v8h a1 = *(const v8h*)(arow + s * 32 + 16);
      afrag[s] = __builtin_shufflevector(a0, a1, 0, 1, 2, 3, 4, 5, 6, 7,
                                         8, 9, 10, 11, 12, 13, 14, 15);
    }
  }

#pragma unroll 1
  for (int t = wave; t < NT; t += 8) {
    const int nt = t >> 2;
    const int nG = (nt << 4) + l16;

    // Prefetch next col-tile's weight row while this tile computes.
    if (t + 8 < NT) {
      const int nGn = (((t + 8) >> 2) << 4) + l16;
      __builtin_prefetch(WT + (size_t)nGn * K + hi * 16, 0, 0);
    }

    const float bv = bias[nG];
    v8f acc;
#pragma unroll
    for (int i = 0; i < 8; ++i) acc[i] = bv;

    // B: lane holds col l16 of tile nt, K {kh..kh+7, kh+16..kh+23}, kh = hi*8.
    const f16_t* brow = WT + (size_t)nG * K + hi * 8;
#pragma unroll
    for (int s = 0; s < KSTEPS; ++s) {
      v8h b0 = *(const v8h*)(brow + s * 32);
      v8h b1 = *(const v8h*)(brow + s * 32 + 16);
      v16h B = __builtin_shufflevector(b0, b1, 0, 1, 2, 3, 4, 5, 6, 7,
                                       8, 9, 10, 11, 12, 13, 14, 15);
      acc = __builtin_amdgcn_wmma_f32_16x16x32_f16(
          false, afrag[s], false, B, (short)0, acc, false, false);
    }

    // D: vgpr v -> row (v + 8*hi) of tile, col l16.  ReLU + f16 store.
    const int mBase = (mt << 4) + (hi << 3);
#pragma unroll
    for (int v = 0; v < 8; ++v) {
      float x = acc[v];
      x = x > 0.f ? x : 0.f;
      Xout[(mBase + v) * N + nG] = (f16_t)x;
    }
  }
}

// ---------------------------------------------------------------------------
// K3: gather + MLP + maxpool + projection.  grid = B*NGRP, block = 256.
// Dynamic LDS: feats 4KB | bufA 32KB | bufB 64KB | pooled 2KB = 104448 B.
// ---------------------------------------------------------------------------
__global__ void group_mlp_kernel(
    const float* __restrict__ xyz, const float* __restrict__ color,
    const float* __restrict__ centers, const int* __restrict__ idx,
    const f16_t* __restrict__ W1T, const float* __restrict__ b1,
    const f16_t* __restrict__ W2T, const float* __restrict__ b2,
    const f16_t* __restrict__ W3T, const float* __restrict__ b3,
    const f16_t* __restrict__ W4T, const float* __restrict__ b4,
    const float* __restrict__ Wp, const float* __restrict__ bp,
    float* __restrict__ emb) {
  extern __shared__ char smem[];
  f16_t* feats  = (f16_t*)smem;                       // 64 x 32
  f16_t* bufA   = (f16_t*)(smem + 4096);              // up to 64 x 256
  f16_t* bufB   = (f16_t*)(smem + 4096 + 32768);      // up to 64 x 512
  float* pooled = (float*)(smem + 4096 + 32768 + 65536);  // 512 f32

  const int g   = blockIdx.x;
  const int b   = g >> 9;
  const int tid = threadIdx.x;

  // Gather neighborhood (centered xyz) + color; pad K 6 -> 32 with zeros.
  if (tid < GSZ) {
    const int i = idx[(size_t)g * GSZ + tid];
    const float* p  = xyz   + ((size_t)b * NPTS + i) * 3;
    const float* cl = color + ((size_t)b * NPTS + i) * 3;
    const float* cc = centers + (size_t)g * 3;
    f16_t* fr = feats + tid * 32;
    fr[0] = (f16_t)(p[0] - cc[0]);
    fr[1] = (f16_t)(p[1] - cc[1]);
    fr[2] = (f16_t)(p[2] - cc[2]);
    fr[3] = (f16_t)cl[0];
    fr[4] = (f16_t)cl[1];
    fr[5] = (f16_t)cl[2];
#pragma unroll
    for (int k = 6; k < 32; ++k) fr[k] = (f16_t)0.f;
  }
  __syncthreads();

  mlp_layer<32, 64>(feats, W1T, b1, bufA);
  __syncthreads();
  mlp_layer<64, 128>(bufA, W2T, b2, bufB);
  __syncthreads();
  mlp_layer<128, 256>(bufB, W3T, b3, bufA);
  __syncthreads();
  mlp_layer<256, 512>(bufA, W4T, b4, bufB);
  __syncthreads();

  // Max over the 64 group points (all values already ReLU'd, >= 0).
  for (int c = tid; c < 512; c += 256) {
    float m = 0.f;
    for (int r = 0; r < GSZ; ++r) {
      float x = (float)bufB[r * 512 + c];
      m = x > m ? x : m;
    }
    pooled[c] = m;
  }
  __syncthreads();

  // Projection 512 -> 60 (fp32, tiny).
  if (tid < EMB) {
    float s = bp[tid];
    for (int k = 0; k < 512; ++k) s += pooled[k] * Wp[k * EMB + tid];
    emb[(size_t)g * EMB + tid] = s;
  }
}

// ---------------------------------------------------------------------------
extern "C" void kernel_launch(void* const* d_in, const int* in_sizes, int n_in,
                              void* d_out, int out_size, void* d_ws,
                              size_t ws_size, hipStream_t stream) {
  (void)in_sizes; (void)n_in; (void)out_size; (void)ws_size;

  const float* xyz   = (const float*)d_in[0];
  const float* color = (const float*)d_in[1];
  const float* W1 = (const float*)d_in[2];  const float* b1 = (const float*)d_in[3];
  const float* W2 = (const float*)d_in[4];  const float* b2 = (const float*)d_in[5];
  const float* W3 = (const float*)d_in[6];  const float* b3 = (const float*)d_in[7];
  const float* W4 = (const float*)d_in[8];  const float* b4 = (const float*)d_in[9];
  const float* Wp = (const float*)d_in[10]; const float* bp = (const float*)d_in[11];

  float* emb     = (float*)d_out;                 // [4*512*60]
  float* centers = emb + 4 * NGRP * EMB;          // [4*512*3]

  char*  ws  = (char*)d_ws;
  int*   knn = (int*)ws;                          // 4*512*64*4 = 524288 B
  f16_t* W1T = (f16_t*)(ws + 524288);             //  64 x  32 f16 =   4096 B
  f16_t* W2T = (f16_t*)(ws + 528384);             // 128 x  64 f16 =  16384 B
  f16_t* W3T = (f16_t*)(ws + 544768);             // 256 x 128 f16 =  65536 B
  f16_t* W4T = (f16_t*)(ws + 610304);             // 512 x 256 f16 = 262144 B

  cvt_wT<<<(64 * 32 + 255) / 256, 256, 0, stream>>>(W1, W1T, 6, 64, 32);
  cvt_wT<<<(128 * 64 + 255) / 256, 256, 0, stream>>>(W2, W2T, 64, 128, 64);
  cvt_wT<<<(256 * 128 + 255) / 256, 256, 0, stream>>>(W3, W3T, 128, 256, 128);
  cvt_wT<<<(512 * 256 + 255) / 256, 256, 0, stream>>>(W4, W4T, 256, 512, 256);

  fps_kernel<<<4, 1024, 0, stream>>>(xyz, centers);

  knn_kernel<<<4 * NGRP, 256, NPTS * 4 + 256 * 8, stream>>>(xyz, centers, knn);

  group_mlp_kernel<<<4 * NGRP, 256, 104448, stream>>>(
      xyz, color, centers, knn, W1T, b1, W2T, b2, W3T, b3, W4T, b4, Wp, bp,
      emb);
}
